// GCN_9328668967072
// MI455X (gfx1250) — matrix operand order repaired and checked by reading, not compiled, not run on hardware
//
#include <hip/hip_runtime.h>

typedef __attribute__((ext_vector_type(16))) _Float16 v16h;
typedef __attribute__((ext_vector_type(8)))  _Float16 v8h;
typedef __attribute__((ext_vector_type(8)))  float    v8f;

#define HDIM 64
#define WT_STRIDE 72   // padded LDS stride (halves): 144B, 16B-aligned, conflict-free

__device__ __forceinline__ void atomAddF(float* p, float v) {
  unsafeAtomicAdd(p, v);   // native global_atomic_add_f32 on gfx1250
}

// ---------------- utility kernels ----------------
__global__ void k_fill(float* __restrict__ p, float v, int n) {
  int i = blockIdx.x * blockDim.x + threadIdx.x;
  if (i < n) p[i] = v;
}

__global__ void k_deg_count(const int* __restrict__ ei, int E, float* __restrict__ deg) {
  int e = blockIdx.x * blockDim.x + threadIdx.x;
  if (e < E) atomAddF(&deg[ei[E + e]], 1.0f);   // dst degrees (self-loop pre-seeded as 1.0)
}

__global__ void k_rsqrt_inplace(float* __restrict__ d, int n) {
  int i = blockIdx.x * blockDim.x + threadIdx.x;
  if (i < n) { float v = d[i]; d[i] = (v > 0.f) ? rsqrtf(v) : 0.f; }
}

__global__ void k_edge_norm(const int* __restrict__ ei, const float* __restrict__ dinv,
                            int E, float* __restrict__ nrm) {
  int e = blockIdx.x * blockDim.x + threadIdx.x;
  if (e < E) nrm[e] = dinv[ei[e]] * dinv[ei[E + e]];
}

// ---------------- WMMA GEMM: T = act(in [+bias]) @ W ; AGG = T * dinv^2 ----------------
// Compile-time KIN / BIAS so all predicates fold; W staged transposed in LDS so each
// B fragment is two ds_load_b128; A fragment is 4 unconditional float4 loads (KIN=64).
template <int KIN, bool BIAS>
__global__ __launch_bounds__(256)
void k_gcn_gemm(const float* __restrict__ in, const float* __restrict__ bias,
                const float* __restrict__ W, const float* __restrict__ dinv,
                float* __restrict__ T, float* __restrict__ AGG, int n_nodes) {
  constexpr int KPAD = (KIN + 31) & ~31;
  constexpr int NKT  = KPAD >> 5;

  __shared__ _Float16 Wt[HDIM * WT_STRIDE];   // transposed: Wt[n][k], 9216 B
  __shared__ float    fb[HDIM];

  const int tid = threadIdx.x;
  // stage W (KIN x 64 row-major) transposed into LDS as f16, zero-padded in K
  for (int idx = tid; idx < KPAD * HDIM; idx += 256) {
    int k = idx >> 6, n = idx & 63;
    Wt[n * WT_STRIDE + k] = (k < KIN) ? (_Float16)W[k * HDIM + n] : (_Float16)0.f;
  }
  if (BIAS && tid < HDIM) fb[tid] = bias[tid];
  __syncthreads();

  const int wave    = tid >> 5;
  const int lane    = tid & 31;
  const int group   = lane >> 4;
  const int rowbase = blockIdx.x * 128 + wave * 16;
  if (rowbase >= n_nodes) return;             // N % 16 == 0: tiles never split

  const int mrow = rowbase + (lane & 15);

  v8f acc[4] = {};
#pragma unroll
  for (int ktile = 0; ktile < NKT; ++ktile) {
    // ---- A fragment (16-bit A 16x32 layout) ----
    v16h a;
    if (KIN == HDIM) {
      // two contiguous 8-float runs per lane: k in [kt*32+g*8, +8) and [kt*32+16+g*8, +8)
      const float* arow = in + (size_t)mrow * HDIM + ktile * 32;
      const float4 r0 = *(const float4*)(arow + group * 8);
      const float4 r1 = *(const float4*)(arow + group * 8 + 4);
      const float4 r2 = *(const float4*)(arow + 16 + group * 8);
      const float4 r3 = *(const float4*)(arow + 16 + group * 8 + 4);
      float4 s0 = {}, s1 = {}, s2 = {}, s3 = {};
      if (BIAS) {
        const float* brow = fb + ktile * 32;
        s0 = *(const float4*)(brow + group * 8);
        s1 = *(const float4*)(brow + group * 8 + 4);
        s2 = *(const float4*)(brow + 16 + group * 8);
        s3 = *(const float4*)(brow + 16 + group * 8 + 4);
      }
#define PUT4(base, r, s)                                              \
      a[base + 0] = (_Float16)(BIAS ? fmaxf(r.x + s.x, 0.f) : r.x);   \
      a[base + 1] = (_Float16)(BIAS ? fmaxf(r.y + s.y, 0.f) : r.y);   \
      a[base + 2] = (_Float16)(BIAS ? fmaxf(r.z + s.z, 0.f) : r.z);   \
      a[base + 3] = (_Float16)(BIAS ? fmaxf(r.w + s.w, 0.f) : r.w);
      PUT4(0,  r0, s0) PUT4(4,  r1, s1) PUT4(8,  r2, s2) PUT4(12, r3, s3)
#undef PUT4
    } else {
      // narrow input (layer 0): clamp address, mask value — no divergent EXEC
      const float* arow = in + (size_t)mrow * KIN;
#pragma unroll
      for (int i = 0; i < 16; ++i) {
        int klocal = group * 8 + i + ((i < 8) ? 0 : 8);
        int k = ktile * 32 + klocal;
        int ksafe = (k < KIN) ? k : (KIN - 1);
        float val = arow[ksafe];
        a[i] = (_Float16)((k < KIN) ? val : 0.f);
      }
    }
    // ---- 4 N-tiles: B fragment = two 8-half runs per lane at k g*8 and 16+g*8 (2x ds_load_b128) ----
#pragma unroll
    for (int nt = 0; nt < 4; ++nt) {
      const int ncol = nt * 16 + (lane & 15);
      const v8h* pb = (const v8h*)&Wt[ncol * WT_STRIDE + ktile * 32 + group * 8];
      const v8h blo = pb[0];
      const v8h bhi = pb[2];
      const v16h b = __builtin_shufflevector(blo, bhi, 0, 1, 2, 3, 4, 5, 6, 7,
                                             8, 9, 10, 11, 12, 13, 14, 15);
      acc[nt] = __builtin_amdgcn_wmma_f32_16x16x32_f16(
          false, a, false, b, (short)0, acc[nt], false, false);
    }
  }

  // ---- epilogue: C/D layout lanes0-15 M=r, lanes16-31 M=8+r ----
  float sn[8];
#pragma unroll
  for (int r = 0; r < 8; ++r) {
    float d = dinv[rowbase + group * 8 + r];
    sn[r] = d * d;                              // self-loop norm = 1/deg
  }
#pragma unroll
  for (int nt = 0; nt < 4; ++nt) {
    const int col = nt * 16 + (lane & 15);
#pragma unroll
    for (int r = 0; r < 8; ++r) {
      const int m = rowbase + group * 8 + r;
      float v = acc[nt][r];
      T  [(size_t)m * HDIM + col] = v;
      AGG[(size_t)m * HDIM + col] = v * sn[r];  // init AGG with self-loop term
    }
  }
}

// ---------------- edge scatter: AGG[dst] += norm[e] * T[src], 16 threads/edge x float4 ----
__global__ void k_edge_scatter(const int* __restrict__ ei, const float* __restrict__ nrm,
                               const float* __restrict__ T, float* __restrict__ AGG, int E) {
  int gid = blockIdx.x * blockDim.x + threadIdx.x;
  int e = gid >> 4;
  if (e >= E) return;
  int c = (gid & 15) << 2;
  int s = ei[e], d = ei[E + e];
  float nm = nrm[e];
  const float4 t = *(const float4*)(T + (size_t)s * HDIM + c);
  float* out = AGG + (size_t)d * HDIM + c;
  atomAddF(out + 0, t.x * nm);
  atomAddF(out + 1, t.y * nm);
  atomAddF(out + 2, t.z * nm);
  atomAddF(out + 3, t.w * nm);
}

// ---------------- pooling: sums[g] += relu(AGG3 + b3) ----------------
__global__ void k_pool_sums(const float* __restrict__ AGG, const float* __restrict__ bias,
                            const int* __restrict__ batch, float* __restrict__ sums, int n_nodes) {
  int gid = blockIdx.x * blockDim.x + threadIdx.x;
  int n = gid >> 4;
  if (n >= n_nodes) return;
  int c = (gid & 15) << 2;
  int g = batch[n];
  const float4 a = *(const float4*)(AGG + (size_t)n * HDIM + c);
  const float4 b = *(const float4*)(bias + c);
  float* out = sums + (size_t)g * HDIM + c;
  atomAddF(out + 0, fmaxf(a.x + b.x, 0.f));
  atomAddF(out + 1, fmaxf(a.y + b.y, 0.f));
  atomAddF(out + 2, fmaxf(a.z + b.z, 0.f));
  atomAddF(out + 3, fmaxf(a.w + b.w, 0.f));
}

__global__ void k_counts(const int* __restrict__ batch, float* __restrict__ counts, int n_nodes) {
  int n = blockIdx.x * blockDim.x + threadIdx.x;
  if (n < n_nodes) atomAddF(&counts[batch[n]], 1.0f);
}

// ---------------- finalize: pooled = sums/max(cnt,1); out = pooled @ Wout + bout ----------
__global__ void k_finalize(const float* __restrict__ sums, const float* __restrict__ counts,
                           const float* __restrict__ Wout, const float* __restrict__ bout,
                           float* __restrict__ out, float* __restrict__ pooled) {
  __shared__ float red[HDIM];
  int g = blockIdx.x, f = threadIdx.x;           // 64 threads = 2 waves
  float cnt = fmaxf(counts[g], 1.0f);
  float p = sums[(size_t)g * HDIM + f] / cnt;
  pooled[(size_t)g * HDIM + f] = p;
  red[f] = p * Wout[f];
  __syncthreads();
  for (int s = 32; s > 0; s >>= 1) {
    if (f < s) red[f] += red[f + s];
    __syncthreads();
  }
  if (f == 0) out[g] = red[0] + bout[0];
}

// ---------------- launcher ----------------
extern "C" void kernel_launch(void* const* d_in, const int* in_sizes, int n_in,
                              void* d_out, int out_size, void* d_ws, size_t ws_size,
                              hipStream_t stream) {
  const float* x     = (const float*)d_in[0];
  const int*   ei    = (const int*)  d_in[1];
  const int*   batch = (const int*)  d_in[2];
  const float* W0 = (const float*)d_in[3];  const float* b0 = (const float*)d_in[4];
  const float* W1 = (const float*)d_in[5];  const float* b1 = (const float*)d_in[6];
  const float* W2 = (const float*)d_in[7];  const float* b2 = (const float*)d_in[8];
  const float* W3 = (const float*)d_in[9];  const float* b3 = (const float*)d_in[10];
  const float* Wout = (const float*)d_in[11]; const float* bout = (const float*)d_in[12];

  const int N = in_sizes[2];            // batch_index length = n_nodes
  const int E = in_sizes[1] / 2;
  const int G = out_size / (HDIM + 1);  // out (G) + pooled (G*64)

  // workspace carve (256B aligned)
  char* ws = (char*)d_ws;
  size_t off = 0;
  auto carve = [&](size_t bytes) -> void* {
    void* p = ws + off;
    off += (bytes + 255) & ~(size_t)255;
    return p;
  };
  float* T    = (float*)carve((size_t)N * HDIM * 4);
  float* AGGA = (float*)carve((size_t)N * HDIM * 4);
  float* AGGB = (float*)carve((size_t)N * HDIM * 4);
  float* dinv = (float*)carve((size_t)N * 4);
  float* nrm  = (float*)carve((size_t)E * 4);
  float* sums = (float*)carve((size_t)G * (HDIM + 1) * 4);
  float* counts = sums + (size_t)G * HDIM;

  const int TB = 256;
  auto blks = [](long long n, int tb) { return (unsigned)((n + tb - 1) / tb); };

  // --- edge norms: deg = 1 (self) + indegree; dinv = rsqrt(deg); norm[e] = dinv[s]*dinv[d]
  k_fill         <<<blks(N, TB), TB, 0, stream>>>(dinv, 1.0f, N);
  k_deg_count    <<<blks(E, TB), TB, 0, stream>>>(ei, E, dinv);
  k_rsqrt_inplace<<<blks(N, TB), TB, 0, stream>>>(dinv, N);
  k_edge_norm    <<<blks(E, TB), TB, 0, stream>>>(ei, dinv, E, nrm);
  k_fill         <<<blks((long long)G * (HDIM + 1), TB), TB, 0, stream>>>(sums, 0.f, G * (HDIM + 1));

  const unsigned gemm_grid = blks(N, 128);             // 8 waves x 16 rows per block
  const unsigned edge_grid = blks((long long)E * 16, TB);

  // layer 0: x(N x 9) @ W0
  k_gcn_gemm<9, false><<<gemm_grid, TB, 0, stream>>>(x, nullptr, W0, dinv, T, AGGA, N);
  k_edge_scatter<<<edge_grid, TB, 0, stream>>>(ei, nrm, T, AGGA, E);
  // layer 1: relu(AGGA + b0) @ W1
  k_gcn_gemm<HDIM, true><<<gemm_grid, TB, 0, stream>>>(AGGA, b0, W1, dinv, T, AGGB, N);
  k_edge_scatter<<<edge_grid, TB, 0, stream>>>(ei, nrm, T, AGGB, E);
  // layer 2
  k_gcn_gemm<HDIM, true><<<gemm_grid, TB, 0, stream>>>(AGGB, b1, W2, dinv, T, AGGA, N);
  k_edge_scatter<<<edge_grid, TB, 0, stream>>>(ei, nrm, T, AGGA, E);
  // layer 3
  k_gcn_gemm<HDIM, true><<<gemm_grid, TB, 0, stream>>>(AGGA, b2, W3, dinv, T, AGGB, N);
  k_edge_scatter<<<edge_grid, TB, 0, stream>>>(ei, nrm, T, AGGB, E);

  // pooling (fused bias+relu of layer 3) and head
  k_pool_sums<<<blks((long long)N * 16, TB), TB, 0, stream>>>(AGGB, b3, batch, sums, N);
  k_counts   <<<blks(N, TB), TB, 0, stream>>>(batch, counts, N);

  float* outp   = (float*)d_out;        // [G] then pooled [G*64]
  float* pooled = outp + G;
  k_finalize<<<G, HDIM, 0, stream>>>(sums, counts, Wout, bout, outp, pooled);
}
